// FlaxGemmaAttention_87445534147015
// MI455X (gfx1250) — matrix operand order repaired and checked by reading, not compiled, not run on hardware
//
#include <hip/hip_runtime.h>
#include <hip/hip_bf16.h>
#include <cstdint>

#define BB  2
#define SS  2048
#define DD  2048
#define HH  8
#define HDD 256

typedef __attribute__((ext_vector_type(16))) __bf16       bf16x16;
typedef __attribute__((ext_vector_type(8)))  float        f32x8;
typedef __attribute__((ext_vector_type(4)))  unsigned int u32x4;
typedef __attribute__((ext_vector_type(8)))  unsigned int u32x8;
typedef __attribute__((ext_vector_type(4)))  int          i32x4;
typedef __attribute__((ext_vector_type(8)))  int          i32x8;

__device__ __forceinline__ unsigned short f2bf(float f) {
  unsigned int u = __builtin_bit_cast(unsigned int, f);
  u += 0x7FFFu + ((u >> 16) & 1u);           // round-to-nearest-even
  return (unsigned short)(u >> 16);
}

__device__ __forceinline__ u32x8 cat8(u32x4 lo, u32x4 hi) {
  u32x8 r;
  r[0] = lo[0]; r[1] = lo[1]; r[2] = lo[2]; r[3] = lo[3];
  r[4] = hi[0]; r[5] = hi[1]; r[6] = hi[2]; r[7] = hi[3];
  return r;
}

__device__ __forceinline__ f32x8 wmma_bf16(u32x8 a, u32x8 b, f32x8 c) {
  return __builtin_amdgcn_wmma_f32_16x16x32_bf16(
      false, __builtin_bit_cast(bf16x16, a),
      false, __builtin_bit_cast(bf16x16, b),
      (short)0, c, false, false);
}

// ---- Tensor Data Mover: 2D tile (tile_rows x row_units of 4B) global->LDS ----
// D# layout per CDNA5 ISA; tensor dims == tile dims so
// the OOB machinery is inert (all our tiles are exact).
__device__ __forceinline__ void tdm_load_2d(unsigned lds_off, const void* gptr,
                                            unsigned tile_rows, unsigned row_units,
                                            unsigned stride_units) {
  unsigned long long ga = (unsigned long long)(size_t)gptr;
  u32x4 g0;
  g0[0] = 1u;                                        // count=1, user descriptor
  g0[1] = lds_off;                                   // lds_addr (bytes)
  g0[2] = (unsigned)(ga & 0xFFFFFFFFu);              // global_addr[31:0]
  g0[3] = (unsigned)((ga >> 32) & 0x01FFFFFFu) | (2u << 30);  // addr[56:32] | type=2
  i32x8 g1;
  g1[0] = (int)(2u << 16);                           // wg_mask=0, data_size=2 (4B)
  g1[1] = (int)(row_units << 16);                    // tensor_dim0 lo16 (hi16=0)
  g1[2] = (int)(tile_rows << 16);                    // tensor_dim0 hi | tensor_dim1 lo
  g1[3] = (int)(row_units << 16);                    // tensor_dim1 hi | tile_dim0
  g1[4] = (int)tile_rows;                            // tile_dim1 (tile_dim2=0)
  g1[5] = (int)stride_units;                         // tensor_dim0_stride lo32
  g1[6] = 0;                                         // stride hi | dim1_stride lo
  g1[7] = 0;
  i32x4 z4 = {0, 0, 0, 0};
#if defined(__clang_major__) && __clang_major__ >= 23
  i32x8 z8 = {0, 0, 0, 0, 0, 0, 0, 0};
  __builtin_amdgcn_tensor_load_to_lds(g0, g1, z4, z4, z8, 0);
#else
  __builtin_amdgcn_tensor_load_to_lds(g0, g1, z4, z4, 0);
#endif
}

// ---------------- fp32 -> bf16 cast ----------------
__global__ void cast_bf16_k(const float* __restrict__ x,
                            unsigned short* __restrict__ y, long n) {
  long i = (long)blockIdx.x * blockDim.x + threadIdx.x;
  if (i < n) y[i] = f2bf(x[i]);
}

// ---------------- fp32 W[K][N] -> bf16 Wt[N][K] (one-time weight transpose) ----
__global__ void castT_bf16_k(const float* __restrict__ x,
                             unsigned short* __restrict__ y, int K, int N) {
  long i = (long)blockIdx.x * blockDim.x + threadIdx.x;
  if (i < (long)K * N) {
    int n = (int)(i % N), k = (int)(i / N);
    y[(size_t)n * K + k] = f2bf(x[i]);
  }
}

// ---------------- bf16 GEMM: C[M,N] f32 = A[M,K] @ Bt[N,K]^T ----------------
// block = 128 thr (4 waves), tile 64(M) x 128(N), k-step 32.
// TDM double-buffered staging: wave0 issues tensor_load_to_lds for tile i+1,
// waits tensorcnt for tile i, DMA overlaps the 32 WMMAs of the block.
__global__ __launch_bounds__(128) void gemm_bf16_k(
    const unsigned short* __restrict__ A, const unsigned short* __restrict__ Bt,
    float* __restrict__ C, int M, int N, int K) {
  __shared__ __align__(64) unsigned short sA[2][64][32];    // 2 x 4 KB
  __shared__ __align__(64) unsigned short sB[2][128][32];   // 2 x 8 KB, sB[n][k]
  const int tid = threadIdx.x, lane = tid & 31, wv = tid >> 5;
  const int l15 = lane & 15, kh = lane >> 4;
  const int tN = blockIdx.x * 128, tM = blockIdx.y * 64;
  const int nk = K >> 5;
  const unsigned ku = (unsigned)(K >> 1);            // K in 4-byte units

  f32x8 acc[8];
#pragma unroll
  for (int g = 0; g < 8; ++g) acc[g] = f32x8{};

  if (wv == 0) {   // prologue: stage tile 0
    tdm_load_2d((unsigned)(size_t)&sA[0][0][0], A + (size_t)tM * K, 64u, 16u, ku);
    tdm_load_2d((unsigned)(size_t)&sB[0][0][0], Bt + (size_t)tN * K, 128u, 16u, ku);
  }

  for (int i = 0; i < nk; ++i) {
    const int buf = i & 1;
    if (wv == 0) {
      if (i + 1 < nk) {   // stage next tile into other buffer, then wait current
        tdm_load_2d((unsigned)(size_t)&sA[buf ^ 1][0][0],
                    A + (size_t)tM * K + (i + 1) * 32, 64u, 16u, ku);
        tdm_load_2d((unsigned)(size_t)&sB[buf ^ 1][0][0],
                    Bt + (size_t)tN * K + (i + 1) * 32, 128u, 16u, ku);
        __builtin_amdgcn_s_wait_tensorcnt(2);
      } else {
        __builtin_amdgcn_s_wait_tensorcnt(0);
      }
    }
    __syncthreads();   // tile i resident for all waves

    u32x4 alo = *(const u32x4*)(&sA[buf][wv * 16 + l15][16 * kh]);
    u32x4 ahi = *(const u32x4*)(&sA[buf][wv * 16 + l15][16 * kh + 8]);
    u32x8 afr = cat8(alo, ahi);
    u32x8 bfr[8];
#pragma unroll
    for (int g = 0; g < 8; ++g)
      bfr[g] = *(const u32x8*)(&sB[buf][g * 16 + l15][16 * kh]);
#pragma unroll
    for (int g = 0; g < 8; ++g)
      acc[g] = wmma_bf16(afr, bfr[g], acc[g]);
    __syncthreads();   // compute done -> buffer reusable by future DMA
  }

#pragma unroll
  for (int g = 0; g < 8; ++g)
#pragma unroll
    for (int r = 0; r < 8; ++r) {
      int row = tM + wv * 16 + r + 8 * kh;
      int col = tN + g * 16 + l15;
      C[(size_t)row * N + col] = acc[g][r];
    }
}

// ---------------- RoPE (+scale) f32 -> bf16, output head-major [b][h][s][hd] ----------------
__global__ void rope_k(const float* __restrict__ X, const int* __restrict__ pos,
                       unsigned short* __restrict__ Y, int nh, float scale) {
  long idx = (long)blockIdx.x * blockDim.x + threadIdx.x;   // B*S*nh*128 threads
  int  i  = (int)(idx & 127);
  long t  = idx >> 7;
  int  h  = (int)(t % nh);
  long bs = t / nh;                                         // b*S + s
  int  p  = pos[bs];
  float inv = __powf(10000.0f, -(float)i * (1.0f / 128.0f));
  float ang = (float)p * inv;
  float cs = __cosf(ang), sn = __sinf(ang);
  const float* src = X + (bs * nh + h) * (long)HDD;
  float x1 = src[i], x2 = src[i + 128];
  long b = bs / SS, s = bs % SS;
  unsigned short* dst = Y + ((b * nh + h) * (long)SS + s) * HDD;
  dst[i]       = f2bf((x1 * cs - x2 * sn) * scale);
  dst[i + 128] = f2bf((x2 * cs + x1 * sn) * scale);
}

// ---------------- V transpose f32[b][s][hd] -> bf16 Vt[b][hd][s] ----------------
__global__ void vtrans_k(const float* __restrict__ V, unsigned short* __restrict__ Vt) {
  long idx = (long)blockIdx.x * blockDim.x + threadIdx.x;   // B*S*HDD
  int  c  = (int)(idx & 255);
  long bs = idx >> 8;
  long b = bs / SS, s = bs % SS;
  Vt[(b * HDD + c) * (long)SS + s] = f2bf(V[idx]);
}

// ---------------- flash attention ----------------
// grid = B*H*(S/16) blocks, 128 threads (4 waves). Waves split the key
// sequence into 32-key tiles (stride 4); private online-softmax state and a
// 16x256 f32 accumulator per wave; split-K logsumexp merge through LDS.
__global__ __launch_bounds__(128, 1) void attn_k(
    const unsigned short* __restrict__ Qbf,   // [B][H][S][256] (pre-scaled HD^-0.5)
    const unsigned short* __restrict__ Kbf,   // [B][S][256]
    const unsigned short* __restrict__ Vt,    // [B][256][S]
    const int*            __restrict__ am,    // [B][S]
    unsigned short*       __restrict__ Obf) { // [B][S][H*256]
  __shared__ __align__(32) unsigned short lq[16 * HDD];     // 8 KB
  __shared__ __align__(32) unsigned short lp[4][16 * 32];   // 4 KB
  __shared__ float lml[4][16][2];
  __shared__ float lo[16 * HDD];                            // 16 KB merge buffer

  const int tid = threadIdx.x, lane = tid & 31, wv = tid >> 5;
  const int l15 = lane & 15, kh = lane >> 4;
  const int blk = blockIdx.x;
  const int qt = blk & 127, h = (blk >> 7) & 7, b = blk >> 10;
  const int q0 = qt * 16;

  {   // stage Q tile (16 x 256 bf16)
    int row = tid >> 3, c = tid & 7;
    *(u32x8*)(lq + row * HDD + c * 32) =
        *(const u32x8*)(Qbf + ((size_t)(b * HH + h) * SS + q0 + row) * HDD + c * 32);
  }
  __syncthreads();

  f32x8 O[16];
#pragma unroll
  for (int g = 0; g < 16; ++g) O[g] = f32x8{};
  float mrow[8], lrow[8];
#pragma unroll
  for (int r = 0; r < 8; ++r) { mrow[r] = -__builtin_inff(); lrow[r] = 0.0f; }

  const unsigned short* lqrow = lq + l15 * HDD;
  const int nj = (q0 + 16 + 31) >> 5;
  const float NEGF = -3.0e38f;

  for (int jt = wv; jt < nj; jt += 4) {
    const int j0 = jt * 32;
    const unsigned short* kbase = Kbf + ((size_t)b * SS + j0) * HDD;
    f32x8 s0 = f32x8{}, s1 = f32x8{};

#pragma unroll
    for (int c = 0; c < 8; ++c) {   // QK^T over HD in chunks of 32
      u32x4 alo = *(const u32x4*)(lqrow + 32 * c + 16 * kh);
      u32x4 ahi = *(const u32x4*)(lqrow + 32 * c + 16 * kh + 8);
      u32x8 afr = cat8(alo, ahi);
      u32x8 b0 = *(const u32x8*)(kbase + (size_t)l15 * HDD + 32 * c + 16 * kh);
      u32x8 b1 = *(const u32x8*)(kbase + (size_t)(16 + l15) * HDD + 32 * c + 16 * kh);
      s0 = wmma_bf16(afr, b0, s0);
      s1 = wmma_bf16(afr, b1, s1);
    }

    // causal + padding mask (C layout: row = r + 8*kh, col = j0 + l15 (+16))
    const int c0 = j0 + l15, c1 = c0 + 16;
    const int p0 = am[(size_t)b * SS + c0], p1 = am[(size_t)b * SS + c1];
#pragma unroll
    for (int r = 0; r < 8; ++r) {
      int row = q0 + r + 8 * kh;
      if (!(c0 <= row && p0 > 0)) s0[r] = NEGF;
      if (!(c1 <= row && p1 > 0)) s1[r] = NEGF;
    }

    float rmax[8];
#pragma unroll
    for (int r = 0; r < 8; ++r) rmax[r] = fmaxf(s0[r], s1[r]);
#pragma unroll
    for (int mm = 1; mm < 16; mm <<= 1)
#pragma unroll
      for (int r = 0; r < 8; ++r) rmax[r] = fmaxf(rmax[r], __shfl_xor(rmax[r], mm, 32));

    float al[8];
#pragma unroll
    for (int r = 0; r < 8; ++r) {
      float mn = fmaxf(mrow[r], rmax[r]);
      al[r] = __expf(mrow[r] - mn);
      mrow[r] = mn;
    }
#pragma unroll
    for (int r = 0; r < 8; ++r) {
      s0[r] = __expf(s0[r] - mrow[r]);
      s1[r] = __expf(s1[r] - mrow[r]);
    }
    float rs[8];
#pragma unroll
    for (int r = 0; r < 8; ++r) rs[r] = s0[r] + s1[r];
#pragma unroll
    for (int mm = 1; mm < 16; mm <<= 1)
#pragma unroll
      for (int r = 0; r < 8; ++r) rs[r] += __shfl_xor(rs[r], mm, 32);
#pragma unroll
    for (int r = 0; r < 8; ++r) lrow[r] = lrow[r] * al[r] + rs[r];

#pragma unroll
    for (int g = 0; g < 16; ++g)
#pragma unroll
      for (int r = 0; r < 8; ++r) O[g][r] *= al[r];

    // P (C layout) -> LDS -> A-fragment layout
    unsigned short* pw = lp[wv];
#pragma unroll
    for (int r = 0; r < 8; ++r) {
      int row = r + 8 * kh;
      pw[row * 32 + l15]      = f2bf(s0[r]);
      pw[row * 32 + 16 + l15] = f2bf(s1[r]);
    }
    u32x4 plo = *(const u32x4*)(pw + l15 * 32 + 16 * kh);
    u32x4 phi = *(const u32x4*)(pw + l15 * 32 + 16 * kh + 8);
    u32x8 pfr = cat8(plo, phi);

    // O += P @ V  (Vt rows contiguous in token dim)
    const unsigned short* vb = Vt + (size_t)b * HDD * SS + j0 + 16 * kh;
#pragma unroll
    for (int g = 0; g < 16; ++g) {
      u32x8 bfr = *(const u32x8*)(vb + (size_t)(g * 16 + l15) * SS);
      O[g] = wmma_bf16(pfr, bfr, O[g]);
    }
  }

  // ---- split-K merge across the 4 waves ----
  if (l15 == 0) {   // lanes 0 and 16 hold rows 0-7 / 8-15 state
#pragma unroll
    for (int r = 0; r < 8; ++r) {
      lml[wv][r + 8 * kh][0] = mrow[r];
      lml[wv][r + 8 * kh][1] = lrow[r];
    }
  }
  __syncthreads();

  float fw[8];
#pragma unroll
  for (int r = 0; r < 8; ++r) {
    int row = r + 8 * kh;
    float M = lml[0][row][0];
    M = fmaxf(M, lml[1][row][0]);
    M = fmaxf(M, lml[2][row][0]);
    M = fmaxf(M, lml[3][row][0]);
    fw[r] = __expf(mrow[r] - M);
  }
#pragma unroll
  for (int g = 0; g < 16; ++g)
#pragma unroll
    for (int r = 0; r < 8; ++r) O[g][r] *= fw[r];

  for (int step = 0; step < 4; ++step) {   // serialized accumulation into lo
    if (wv == step) {
#pragma unroll
      for (int g = 0; g < 16; ++g)
#pragma unroll
        for (int r = 0; r < 8; ++r) {
          int row = r + 8 * kh, col = g * 16 + l15;
          if (step == 0) lo[row * HDD + col] = O[g][r];
          else           lo[row * HDD + col] += O[g][r];
        }
    }
    __syncthreads();
  }

  {   // normalize + store bf16 into attention-output buffer
    int row = tid >> 3, cb = (tid & 7) * 32;
    float M = lml[0][row][0];
    M = fmaxf(M, lml[1][row][0]);
    M = fmaxf(M, lml[2][row][0]);
    M = fmaxf(M, lml[3][row][0]);
    float lt = 0.0f;
#pragma unroll
    for (int w = 0; w < 4; ++w) lt += __expf(lml[w][row][0] - M) * lml[w][row][1];
    float inv = 1.0f / lt;
    unsigned short* dst =
        Obf + ((size_t)b * SS + q0 + row) * (HH * HDD) + h * HDD + cb;
#pragma unroll
    for (int i = 0; i < 32; ++i) dst[i] = f2bf(lo[row * HDD + cb + i] * inv);
  }
}

// ---------------- host-side launch ----------------
extern "C" void kernel_launch(void* const* d_in, const int* in_sizes, int n_in,
                              void* d_out, int out_size, void* d_ws, size_t ws_size,
                              hipStream_t stream) {
  (void)in_sizes; (void)n_in; (void)out_size; (void)ws_size;
  const float* Xf  = (const float*)d_in[0];
  const int*   am  = (const int*)d_in[1];
  const int*   pid = (const int*)d_in[2];
  const float* Wq  = (const float*)d_in[3];
  const float* Wk  = (const float*)d_in[4];
  const float* Wv  = (const float*)d_in[5];
  const float* Wo  = (const float*)d_in[6];
  float* out = (float*)d_out;

  char* ws = (char*)d_ws;
  size_t off = 0;
  auto alloc = [&](size_t bytes) -> void* {
    void* p = ws + off;
    off += (bytes + 255) & ~(size_t)255;
    return p;
  };

  const long MROWS = (long)BB * SS;          // 4096
  unsigned short* Xbf = (unsigned short*)alloc(MROWS * DD * 2);
  unsigned short* Wqb = (unsigned short*)alloc((long)DD * DD * 2);     // [N][K]
  unsigned short* Wkb = (unsigned short*)alloc((long)HDD * DD * 2);    // [N][K]
  unsigned short* Wvb = (unsigned short*)alloc((long)HDD * DD * 2);    // [N][K]
  unsigned short* Wob = (unsigned short*)alloc((long)DD * DD * 2);     // [N][K]
  float* Qf = (float*)alloc(MROWS * (long)(HH * HDD) * 4);
  float* Kf = (float*)alloc(MROWS * (long)HDD * 4);
  float* Vf = (float*)alloc(MROWS * (long)HDD * 4);
  unsigned short* Qbf = (unsigned short*)alloc((long)BB * HH * SS * HDD * 2);
  unsigned short* Kbf = (unsigned short*)alloc((long)BB * SS * HDD * 2);
  unsigned short* Vtb = (unsigned short*)alloc((long)BB * HDD * SS * 2);
  unsigned short* Abf = (unsigned short*)alloc(MROWS * DD * 2);

  // casts (weights transposed once so GEMM B-fragments are contiguous)
  cast_bf16_k<<<dim3((unsigned)((MROWS * DD + 255) / 256)), dim3(256), 0, stream>>>(
      Xf, Xbf, MROWS * DD);
  castT_bf16_k<<<dim3((unsigned)(((long)DD * DD + 255) / 256)), dim3(256), 0, stream>>>(
      Wq, Wqb, DD, DD);
  castT_bf16_k<<<dim3((unsigned)(((long)DD * HDD + 255) / 256)), dim3(256), 0, stream>>>(
      Wk, Wkb, DD, HDD);
  castT_bf16_k<<<dim3((unsigned)(((long)DD * HDD + 255) / 256)), dim3(256), 0, stream>>>(
      Wv, Wvb, DD, HDD);
  castT_bf16_k<<<dim3((unsigned)(((long)DD * DD + 255) / 256)), dim3(256), 0, stream>>>(
      Wo, Wob, DD, DD);

  // projections (TDM-staged WMMA GEMMs)
  gemm_bf16_k<<<dim3(DD / 128, MROWS / 64), dim3(128), 0, stream>>>(
      Xbf, Wqb, Qf, (int)MROWS, DD, DD);
  gemm_bf16_k<<<dim3(HDD / 128, MROWS / 64), dim3(128), 0, stream>>>(
      Xbf, Wkb, Kf, (int)MROWS, HDD, DD);
  gemm_bf16_k<<<dim3(HDD / 128, MROWS / 64), dim3(128), 0, stream>>>(
      Xbf, Wvb, Vf, (int)MROWS, HDD, DD);

  // RoPE (HD^-0.5 folded into Q) + V transpose
  {
    long tq = (long)BB * SS * HH * 128;
    rope_k<<<dim3((unsigned)(tq / 256)), dim3(256), 0, stream>>>(Qf, pid, Qbf, HH, 0.0625f);
    long tk = (long)BB * SS * 1 * 128;
    rope_k<<<dim3((unsigned)(tk / 256)), dim3(256), 0, stream>>>(Kf, pid, Kbf, 1, 1.0f);
    long tv = (long)BB * SS * HDD;
    vtrans_k<<<dim3((unsigned)(tv / 256)), dim3(256), 0, stream>>>(Vf, Vtb);
  }

  // fused flash attention
  attn_k<<<dim3(BB * HH * (SS / 16)), dim3(128), 0, stream>>>(Qbf, Kbf, Vtb, am, Abf);

  // output projection -> f32 result
  gemm_bf16_k<<<dim3(DD / 128, MROWS / 64), dim3(128), 0, stream>>>(
      Abf, Wob, out, (int)MROWS, DD, DD);
}
